// SSIMCustom_47021301956631
// MI455X (gfx1250) — matrix operand order repaired and checked by reading, not compiled, not run on hardware
//
#include <hip/hip_runtime.h>

typedef __attribute__((ext_vector_type(16))) _Float16 v16h;
typedef __attribute__((ext_vector_type(8)))  _Float16 v8h;
typedef __attribute__((ext_vector_type(8)))  float    v8f;

#define WSZ   11
#define PAD   5
#define TILE  16
#define XDIM  32      // padded X tile (26 used, rest zero)
#define HROWS 26      // valid rows of intermediate H

// Normalized Gaussian tap for offset index d in [0,10], sigma = 1.5.
// 1/sum(exp(-(i-5)^2/4.5)) = 0.2660117
__device__ __forceinline__ float gw(int d) {
    float o = (float)(d - PAD);
    return __expf(-o * o * (1.0f / 4.5f)) * 0.2660117f;
}

// One wave32 per 16x16 output tile of one (n,c) plane.
// Separable 11-tap Gaussian conv as banded-matrix WMMAs:
//   H(26x16) = X(32x32, zero-padded) * Wb   (2 row-chunks x 5 fields = 10 wmma)
//   V(16x16) = Wv * H(32x16, zero-padded)   (5 wmma)
// All LDS buffers are padded to 32 so no access needs a bounds guard:
// zero-padded A rows/cols make the unused D rows exactly zero.
__launch_bounds__(32)
__global__ void ssim_tile_wmma(const float* __restrict__ img1,
                               const float* __restrict__ img2,
                               float* __restrict__ sink,   // partials array or scalar out
                               int H, int W, int mode, float scale)
{
    __shared__ __align__(32) float    X1[XDIM * XDIM];
    __shared__ __align__(32) float    X2[XDIM * XDIM];
    // column-major: Hb[f][col*32 + row], 32 rows per column (rows >=26 are zero)
    __shared__ __align__(32) _Float16 Hb[5][TILE * XDIM];

    const int tid  = threadIdx.x;
    const int hb   = tid >> 4;     // half-wave select
    const int lnlo = tid & 15;

    const int tilesX        = W / TILE;
    const int tilesPerPlane = (H / TILE) * tilesX;
    const int plane = blockIdx.x / tilesPerPlane;
    const int t     = blockIdx.x % tilesPerPlane;
    const int row0  = (t / tilesX) * TILE;
    const int col0  = (t % tilesX) * TILE;
    const float* p1 = img1 + (size_t)plane * H * W;
    const float* p2 = img2 + (size_t)plane * H * W;

    // ---- Banded filter operands (per-wave constants) ----
    // Wb in B layout (32x16): B[K][N] = w[K-N], K = (e&7)+8*hb+16*(e>>3), N = lnlo
    v16h wB;
    #pragma unroll
    for (int e = 0; e < 16; e++) {
        int d = (e & 7) + 8 * hb + 16 * (e >> 3) - lnlo;
        wB[e] = (_Float16)((d >= 0 && d < WSZ) ? gw(d) : 0.0f);
    }
    // Wv in A layout (16x32): A[M][K] = w[K-M], M = lnlo, K = e + 16*hb
    v16h wA;
    #pragma unroll
    for (int e = 0; e < 16; e++) {
        int d = e + 16 * hb - lnlo;
        wA[e] = (_Float16)((d >= 0 && d < WSZ) ? gw(d) : 0.0f);
    }

    // ---- Stage zero-padded 32x32 input windows (branch-free: clamp + select) ----
    for (int it = 0; it < XDIM; it++) {
        int idx = it * XDIM + tid;        // row = it, col = tid (coalesced)
        int r = it, c = tid;
        int gr = row0 + r - PAD, gc = col0 + c - PAD;
        bool ok = (r < HROWS) & (c < HROWS) &
                  (gr >= 0) & (gr < H) & (gc >= 0) & (gc < W);
        int grc = min(max(gr, 0), H - 1);
        int gcc = min(max(gc, 0), W - 1);
        float a = p1[(size_t)grc * W + gcc];
        float b = p2[(size_t)grc * W + gcc];
        X1[idx] = ok ? a : 0.0f;
        X2[idx] = ok ? b : 0.0f;
    }
    __syncthreads();

    // ---- Horizontal pass: H[r][j] = sum_b w[b] * X[r][j+b] ----
    #pragma unroll
    for (int ch = 0; ch < 2; ch++) {
        const int rowA = lnlo + ch * 16;      // X row held by this lane (<=31, in-bounds)
        // A-layout K indices are two contiguous runs of 8: [8*hb, +8) and [16+8*hb, +8)
        v8f xlo1 = *(const v8f*)&X1[rowA * XDIM + 8 * hb];
        v8f xhi1 = *(const v8f*)&X1[rowA * XDIM + 16 + 8 * hb];
        v8f xlo2 = *(const v8f*)&X2[rowA * XDIM + 8 * hb];
        v8f xhi2 = *(const v8f*)&X2[rowA * XDIM + 16 + 8 * hb];

        v16h A[5];
        #pragma unroll
        for (int e = 0; e < 16; e++) {
            float x1 = (e < 8) ? xlo1[e & 7] : xhi1[e & 7];
            float x2 = (e < 8) ? xlo2[e & 7] : xhi2[e & 7];
            A[0][e] = (_Float16)x1;
            A[1][e] = (_Float16)x2;
            A[2][e] = (_Float16)(x1 * x1);
            A[3][e] = (_Float16)(x2 * x2);
            A[4][e] = (_Float16)(x1 * x2);
        }
        #pragma unroll
        for (int f = 0; f < 5; f++) {
            v8f c = {};
            v8f d = __builtin_amdgcn_wmma_f32_16x16x32_f16(
                        false, A[f], false, wB, (short)0, c, false, false);
            // Lane's 8 D values are 8 consecutive H rows of column lnlo: one 16B store.
            v8h hv;
            #pragma unroll
            for (int r8 = 0; r8 < 8; r8++) hv[r8] = (_Float16)d[r8];
            *(v8h*)&Hb[f][lnlo * XDIM + ch * 16 + 8 * hb] = hv;
        }
    }
    __syncthreads();

    // ---- Vertical pass: V[i][j] = sum_a w[a] * H[i+a][j] ----
    v8f V[5];
    #pragma unroll
    for (int f = 0; f < 5; f++) {
        // B element e: K = e + 16*hb -> 16 consecutive rows of column lnlo: one 32B load.
        v16h b = *(const v16h*)&Hb[f][lnlo * XDIM + 16 * hb];
        v8f c = {};
        V[f] = __builtin_amdgcn_wmma_f32_16x16x32_f16(
                   false, wA, false, b, (short)0, c, false, false);
    }

    // ---- SSIM map + wave reduction (8 outputs per lane) ----
    const float C1 = 1e-4f, C2 = 9e-4f;
    float lsum = 0.0f;
    #pragma unroll
    for (int r8 = 0; r8 < 8; r8++) {
        float mu1 = V[0][r8], mu2 = V[1][r8];
        float m11 = mu1 * mu1, m22 = mu2 * mu2, m12 = mu1 * mu2;
        float s1  = V[2][r8] - m11;
        float s2  = V[3][r8] - m22;
        float sx  = V[4][r8] - m12;
        float num = (2.0f * m12 + C1) * (2.0f * sx + C2);
        float den = (m11 + m22 + C1) * (s1 + s2 + C2);
        lsum += num / den;
    }
    #pragma unroll
    for (int off = 16; off > 0; off >>= 1)
        lsum += __shfl_xor(lsum, off, 32);
    if (tid == 0) {
        if (mode == 0) sink[blockIdx.x] = lsum;
        else           atomicAdd(sink, lsum * scale);
    }
}

__global__ void zero_out(float* out) { out[0] = 0.0f; }

__global__ void reduce_partials(const float* __restrict__ partial, int n,
                                float* __restrict__ out, float scale)
{
    __shared__ float sm[256];
    float s = 0.0f;
    for (int i = threadIdx.x; i < n; i += 256) s += partial[i];
    sm[threadIdx.x] = s;
    __syncthreads();
    for (int off = 128; off > 0; off >>= 1) {
        if ((int)threadIdx.x < off) sm[threadIdx.x] += sm[threadIdx.x + off];
        __syncthreads();
    }
    if (threadIdx.x == 0) out[0] = sm[0] * scale;
}

extern "C" void kernel_launch(void* const* d_in, const int* in_sizes, int n_in,
                              void* d_out, int out_size, void* d_ws, size_t ws_size,
                              hipStream_t stream) {
    const float* img1 = (const float*)d_in[0];
    const float* img2 = (const float*)d_in[1];
    float* out = (float*)d_out;

    const int H = 512, W = 512;
    const int planes = in_sizes[0] / (H * W);            // N*C = 96
    const int blocks = planes * (H / TILE) * (W / TILE); // 98304 tiles
    const float scale = 1.0f / (float)in_sizes[0];

    if (ws_size >= (size_t)blocks * sizeof(float)) {
        float* partial = (float*)d_ws;
        ssim_tile_wmma<<<blocks, 32, 0, stream>>>(img1, img2, partial, H, W, 0, 0.0f);
        reduce_partials<<<1, 256, 0, stream>>>(partial, blocks, out, scale);
    } else {
        zero_out<<<1, 1, 0, stream>>>(out);
        ssim_tile_wmma<<<blocks, 32, 0, stream>>>(img1, img2, out, H, W, 1, scale);
    }
}
